// TwoSimplicialAttention_88957362635625
// MI455X (gfx1250) — matrix operand order repaired and checked by reading, not compiled, not run on hardware
//
#include <hip/hip_runtime.h>

// ---------------------------------------------------------------------------
// 2-simplicial attention for MI455X (gfx1250), wave32, WMMA bf16 GEMMs.
// N=8192, IN=OUT=1024, H=8, HD=128, W1=W2=8.
// Round 3: immediate-offset addressing (1 ptr per matrix, 64B increment per
// K-step), compile-time output stride; 64x64 wave tiles, fused 5-proj GEMM.
// ---------------------------------------------------------------------------

#define NROW 8192
#define DIM  1024
#define NH   8
#define HD   128
#define WIN  8
#define NPROJ 5
#define PSTR (NPROJ * DIM)   // 5120: row stride of fused projection output

typedef __attribute__((ext_vector_type(16))) __bf16       v16bf;
typedef __attribute__((ext_vector_type(8)))  float        v8f;
typedef __attribute__((ext_vector_type(4)))  unsigned int u32x4;

union FragAB {
    v16bf v;
    u32x4 q[2];
};

// ---------------------------------------------------------------------------
// f32 -> bf16 conversion (grid-stride)
// ---------------------------------------------------------------------------
__global__ void cvt_f32_bf16_kernel(const float* __restrict__ in,
                                    __bf16* __restrict__ out, int n) {
    int idx = blockIdx.x * blockDim.x + threadIdx.x;
    for (; idx < n; idx += gridDim.x * blockDim.x)
        out[idx] = (__bf16)in[idx];
}

// ---------------------------------------------------------------------------
// Concatenate 5 bias vectors (1024 each) into one 5120 vector.
// ---------------------------------------------------------------------------
__global__ void bias_cat_kernel(const float* __restrict__ b0,
                                const float* __restrict__ b1,
                                const float* __restrict__ b2,
                                const float* __restrict__ b3,
                                const float* __restrict__ b4,
                                float* __restrict__ out) {
    const int i = blockIdx.x * blockDim.x + threadIdx.x;   // 0..5119
    const int p = i >> 10, c = i & (DIM - 1);
    float v = b0[c];
    if (p == 1) v = b1[c];
    else if (p == 2) v = b2[c];
    else if (p == 3) v = b3[c];
    else if (p == 4) v = b4[c];
    if (i < PSTR) out[i] = v;
}

// ---------------------------------------------------------------------------
// C[M, Ncols] = A[M, K=1024] @ W[Ncols, 1024]^T + bias[Ncols]
// Block: 256 thr = 8 waves; block tile 256x128; wave tile 64x64 (4x4 WMMA
// fragments -> 16 v_wmma per 16 global_load_b128).
// One base pointer per matrix; all fragment loads/stores use compile-time
// immediate offsets (fit in the signed 24-bit VMEM IOFFSET); pointers advance
// by 64B per K-step.  OUTSTRIDE is a template constant (no u64 muls).
// ---------------------------------------------------------------------------
template <int OUTSTRIDE>
__global__ __launch_bounds__(256)
void gemm_bf16_wmma_kernel(const __bf16* __restrict__ A,
                           const __bf16* __restrict__ Wt,
                           const float* __restrict__ bias,
                           float* __restrict__ out) {
    const int lane = threadIdx.x & 31;
    const int wave = threadIdx.x >> 5;          // 0..7
    const int wm = wave >> 1;                   // 0..3  (rows of 64)
    const int wn = wave & 1;                    // 0..1  (cols of 64)
    const int rowBase = blockIdx.x * 256 + wm * 64;
    const int colBase = blockIdx.y * 128 + wn * 64;
    const int lr  = lane & 15;                  // row (A) / col (B) within frag
    const int grp = lane >> 4;                  // 0/1: K-half select
    const int kA  = grp * 8;                    // A: K-offset of v0-3 block
    const int kB  = grp * 8;                    // B: K-offset of v0-3 block

    // Single base pointers; mi/ni strides are immediate offsets (16*DIM bf16).
    const __bf16* pA = A  + (size_t)(rowBase + lr) * DIM + kA;
    const __bf16* pB = Wt + (size_t)(colBase + lr) * DIM + kB;

    v8f acc[4][4] = {};

    for (int kt = 0; kt < DIM; kt += 32) {
        FragAB a[4], b[4];
#pragma unroll
        for (int mi = 0; mi < 4; mi++) {
            a[mi].q[0] = *(const u32x4*)(pA + mi * 16 * DIM);        // K kA..kA+7
            a[mi].q[1] = *(const u32x4*)(pA + mi * 16 * DIM + 16);   // K kA+16..+23
        }
#pragma unroll
        for (int ni = 0; ni < 4; ni++) {
            b[ni].q[0] = *(const u32x4*)(pB + ni * 16 * DIM);        // K kB..kB+7
            b[ni].q[1] = *(const u32x4*)(pB + ni * 16 * DIM + 16);   // K kB+16..+23
        }
        if (kt + 32 < DIM) {
            // global_prefetch_b8 of the next A K-tile
            __builtin_prefetch(pA + 32, 0, 1);
        }
#pragma unroll
        for (int mi = 0; mi < 4; mi++)
#pragma unroll
            for (int ni = 0; ni < 4; ni++)
                acc[mi][ni] = __builtin_amdgcn_wmma_f32_16x16x32_bf16(
                    false, a[mi].v, false, b[ni].v,
                    (short)0, acc[mi][ni], false, false);
        pA += 32;
        pB += 32;
    }

    // C layout: VGPR r -> M = r + grp*8, N = lr.
    // Single out base pointer + immediate offsets (max ~1.1 MB < 8 MB).
    float* pC = out + (size_t)(rowBase + grp * 8) * OUTSTRIDE + colBase + lr;
    const float* pBias = bias + colBase + lr;
#pragma unroll
    for (int ni = 0; ni < 4; ni++) {
        const float bv = pBias[ni * 16];
#pragma unroll
        for (int mi = 0; mi < 4; mi++)
#pragma unroll
            for (int r = 0; r < 8; r++)
                pC[(size_t)(mi * 16 + r) * OUTSTRIDE + ni * 16] =
                    acc[mi][ni][r] + bv;
    }
}

// ---------------------------------------------------------------------------
// Windowed triple-product attention. One 64-thread block per (n, h).
// Q/K/V/Kp/Vp live in one fused buffer with row stride PSTR.
// scores[j][k] = <q, k_j, k'_k>/sqrt(HD); joint softmax over 8x8;
// z[d] = sum_{jk} S[jk] * v_j[d] * v'_k[d]. Writes Z as bf16 (stride DIM).
// ---------------------------------------------------------------------------
__global__ __launch_bounds__(64)
void attn_kernel(const float* __restrict__ P, __bf16* __restrict__ Zb) {
    __shared__ float sQ[HD];
    __shared__ float sK[WIN][HD], sKp[WIN][HD], sV[WIN][HD], sVp[WIN][HD];
    __shared__ float sS[WIN * WIN];

    const int i = blockIdx.x;
    const int h = blockIdx.y;
    const int tid = threadIdx.x;                 // 0..63
    const size_t hb = (size_t)h * HD;

    const float* Q  = P + 0 * DIM;
    const float* K  = P + 1 * DIM;
    const float* V  = P + 2 * DIM;
    const float* Kp = P + 3 * DIM;
    const float* Vp = P + 4 * DIM;

    for (int d = tid; d < HD; d += 64)
        sQ[d] = Q[(size_t)i * PSTR + hb + d];
    for (int idx = tid; idx < WIN * HD; idx += 64) {
        const int j = idx >> 7, d = idx & (HD - 1);
        int pos = i - (WIN - 1) + j;
        if (pos < 0) pos = 0;
        const size_t o = (size_t)pos * PSTR + hb + d;
        sK[j][d]  = K[o];  sKp[j][d] = Kp[o];
        sV[j][d]  = V[o];  sVp[j][d] = Vp[o];
    }
    __syncthreads();

    const int j = tid >> 3, k = tid & 7;
    const bool valid = (i - (WIN - 1) + j >= 0) && (i - (WIN - 1) + k >= 0);
    float s = 0.f;
    for (int d = 0; d < HD; d++)
        s += sQ[d] * sK[j][d] * sKp[k][d];
    s *= 0.08838834764831845f;                   // 1/sqrt(128)
    if (!valid) s = -1e30f;
    sS[tid] = s;
    __syncthreads();

    float m = -1e30f;
    for (int t = 0; t < 64; t++) m = fmaxf(m, sS[t]);
    const float e = __expf(s - m);
    __syncthreads();
    sS[tid] = e;
    __syncthreads();
    float sum = 0.f;
    for (int t = 0; t < 64; t++) sum += sS[t];
    const float inv = 1.f / sum;

    const int d0 = tid * 2, d1 = d0 + 1;
    float z0 = 0.f, z1 = 0.f;
    for (int t = 0; t < 64; t++) {
        const int jj = t >> 3, kk = t & 7;
        const float p = sS[t];
        z0 += p * sV[jj][d0] * sVp[kk][d0];
        z1 += p * sV[jj][d1] * sVp[kk][d1];
    }
    Zb[(size_t)i * DIM + hb + d0] = (__bf16)(z0 * inv);
    Zb[(size_t)i * DIM + hb + d1] = (__bf16)(z1 * inv);
}

// ---------------------------------------------------------------------------
// out = LayerNorm(G + x) * gamma + beta, per row of 1024. 256 thr / row.
// ---------------------------------------------------------------------------
__global__ __launch_bounds__(256)
void ln_kernel(const float* __restrict__ G, const float* __restrict__ x,
               const float* __restrict__ gamma, const float* __restrict__ beta,
               float* __restrict__ out) {
    __shared__ float red[256];
    const int row = blockIdx.x, tid = threadIdx.x;
    float v[4];
    float sum = 0.f;
#pragma unroll
    for (int q = 0; q < 4; q++) {
        const int c = tid + q * 256;
        const float t = G[(size_t)row * DIM + c] + x[(size_t)row * DIM + c];
        v[q] = t;
        sum += t;
    }
    red[tid] = sum;
    __syncthreads();
    for (int s = 128; s > 0; s >>= 1) {
        if (tid < s) red[tid] += red[tid + s];
        __syncthreads();
    }
    const float mu = red[0] * (1.f / DIM);
    __syncthreads();
    float sq = 0.f;
#pragma unroll
    for (int q = 0; q < 4; q++) { const float d = v[q] - mu; sq += d * d; }
    red[tid] = sq;
    __syncthreads();
    for (int s = 128; s > 0; s >>= 1) {
        if (tid < s) red[tid] += red[tid + s];
        __syncthreads();
    }
    const float rstd = rsqrtf(red[0] * (1.f / DIM) + 1e-5f);
#pragma unroll
    for (int q = 0; q < 4; q++) {
        const int c = tid + q * 256;
        out[(size_t)row * DIM + c] = (v[q] - mu) * rstd * gamma[c] + beta[c];
    }
}

// ---------------------------------------------------------------------------
// Launch
// in: [0]=x [1]=Wq [2]=bq [3]=Wk [4]=bk [5]=Wv [6]=bv
//     [7]=Wkp [8]=bkp [9]=Wvp [10]=bvp [11]=Wo [12]=bo [13]=gamma [14]=beta
// ---------------------------------------------------------------------------
extern "C" void kernel_launch(void* const* d_in, const int* in_sizes, int n_in,
                              void* d_out, int out_size, void* d_ws, size_t ws_size,
                              hipStream_t stream) {
    const float* x     = (const float*)d_in[0];
    const float* Wsrc[6] = {(const float*)d_in[1], (const float*)d_in[3],
                            (const float*)d_in[5], (const float*)d_in[7],
                            (const float*)d_in[9], (const float*)d_in[11]};
    const float* gamma = (const float*)d_in[13];
    const float* beta  = (const float*)d_in[14];
    const float* bo    = (const float*)d_in[12];

    char* ws = (char*)d_ws;
    size_t off = 0;
    __bf16* xb = (__bf16*)(ws + off); off += (size_t)NROW * DIM * 2;
    // 6 weight matrices, contiguous: first 5 form the fused [5120][1024] matrix
    __bf16* Wb[6];
    for (int i = 0; i < 6; i++) { Wb[i] = (__bf16*)(ws + off); off += (size_t)DIM * DIM * 2; }
    float* bcat = (float*)(ws + off); off += (size_t)PSTR * 4;
    float* P    = (float*)(ws + off); off += (size_t)NROW * PSTR * 4;  // fused Q|K|V|Kp|Vp
    __bf16* Zb  = (__bf16*)(ws + off); off += (size_t)NROW * DIM * 2;
    float*  Gm  = (float*)(ws + off);  off += (size_t)NROW * DIM * 4;

    // 1) bf16 converts + bias concat
    cvt_f32_bf16_kernel<<<2048, 256, 0, stream>>>(x, xb, NROW * DIM);
    for (int i = 0; i < 6; i++)
        cvt_f32_bf16_kernel<<<1024, 256, 0, stream>>>(Wsrc[i], Wb[i], DIM * DIM);
    bias_cat_kernel<<<PSTR / 256, 256, 0, stream>>>(
        (const float*)d_in[2], (const float*)d_in[4], (const float*)d_in[6],
        (const float*)d_in[8], (const float*)d_in[10], bcat);

    // 2) fused projection GEMM: [8192,1024] x [5120,1024]^T -> [8192,5120]
    gemm_bf16_wmma_kernel<PSTR><<<dim3(NROW / 256, PSTR / 128), 256, 0, stream>>>(
        xb, Wb[0], bcat, P);

    // 3) windowed triple-product attention -> Z (bf16)
    attn_kernel<<<dim3(NROW, NH), 64, 0, stream>>>(P, Zb);

    // 4) output projection GEMM: [8192,1024] x [1024,1024]^T -> [8192,1024]
    gemm_bf16_wmma_kernel<DIM><<<dim3(NROW / 256, DIM / 128), 256, 0, stream>>>(
        Zb, Wb[5], bo, Gm);

    // 5) residual + layernorm
    ln_kernel<<<NROW, 256, 0, stream>>>(Gm, x, gamma, beta, (float*)d_out);
}
